// DecoderRNN_38749194944755
// MI455X (gfx1250) — matrix operand order repaired and checked by reading, not compiled, not run on hardware
//
#include <hip/hip_runtime.h>
#include <hip/hip_bf16.h>

typedef _Float16 v8h  __attribute__((ext_vector_type(8)));
typedef _Float16 v16h __attribute__((ext_vector_type(16)));
typedef float    v8f  __attribute__((ext_vector_type(8)));

#define HIDDEN 2048
#define G4     8192      // 4*HIDDEN
#define BATCH  256
#define OUTD   66
#define INDIM  76
#define STEPS  120
#define TFRAME 121       // LENGTH+1

// ---------------------------------------------------------------------------
// f32 -> f16 conversion (grid-stride)
// ---------------------------------------------------------------------------
__global__ void cvt_f32_f16(const float* __restrict__ s, _Float16* __restrict__ d, int n) {
    for (int i = blockIdx.x * blockDim.x + threadIdx.x; i < n; i += gridDim.x * blockDim.x)
        d[i] = (_Float16)s[i];
}

// ---------------------------------------------------------------------------
// x_proj[b,g] = b_ih[g] + b_hh[g] + W_ih[g,66+label] + sum_k inputs[b,k]*W_ih[g,k]
// (one-hot conditioning folds into a single weight-column add)
// ---------------------------------------------------------------------------
__global__ void xproj_kernel(const float* __restrict__ inputs,
                             const long long* __restrict__ labels,
                             const float* __restrict__ W_ih,
                             const float* __restrict__ b_ih,
                             const float* __restrict__ b_hh,
                             float* __restrict__ xproj) {
    int idx = blockIdx.x * blockDim.x + threadIdx.x;
    if (idx >= BATCH * G4) return;
    int b = idx / G4, g = idx % G4;
    int lab = (int)labels[0];
    const float* wr  = W_ih + (long)g * INDIM;
    const float* inp = inputs + (long)b * OUTD;
    float acc = 0.f;
#pragma unroll 6
    for (int k = 0; k < OUTD; ++k) acc += inp[k] * wr[k];
    xproj[idx] = acc + b_ih[g] + b_hh[g] + wr[OUTD + lab];
}

// ---------------------------------------------------------------------------
// h0 = 0 (f16), c = 0, and out[:,0,:] = frame0
// ---------------------------------------------------------------------------
__global__ void init_kernel(const float* __restrict__ inputs,
                            _Float16* __restrict__ h0,
                            float* __restrict__ c,
                            float* __restrict__ out) {
    int i = blockIdx.x * blockDim.x + threadIdx.x;
    if (i < BATCH * HIDDEN) { h0[i] = (_Float16)0.f; c[i] = 0.f; }
    if (i < BATCH * OUTD) {
        int b = i / OUTD, o = i % OUTD;
        out[(long)b * (TFRAME * OUTD) + o] = inputs[i];
    }
}

// ---------------------------------------------------------------------------
// One LSTM step: gates = x_proj + h_in @ W_hh^T, cell update, write h_out/c.
// Wave tile: M=32 batch (2 M-tiles) x N=16 hidden units x 4 gates.
// 8 f32 accumulators, 8 v_wmma_f32_16x16x32_f16 per k-step, K=2048.
// Cell update is fully in-register (each wave owns i,f,g,o of its (b,j) tile).
// ---------------------------------------------------------------------------
__global__ void __launch_bounds__(128)
lstm_step(const _Float16* __restrict__ hin,
          const _Float16* __restrict__ whh,
          const float*    __restrict__ xproj,
          float*          __restrict__ cstate,
          _Float16*       __restrict__ hout) {
    const int lane  = threadIdx.x & 31;
    const int wv    = blockIdx.x * (blockDim.x >> 5) + (threadIdx.x >> 5); // 0..1023
    const int bTile = wv >> 7;      // 0..7   (32 batch rows each)
    const int jTile = wv & 127;     // 0..127 (16 hidden units each)
    const int b0 = bTile * 32;
    const int j0 = jTile * 16;
    const int lhalf = lane >> 4;    // 0/1
    const int lmod  = lane & 15;

    // A fragments (h, 16x32 f16): lane holds row b, K = {koff..koff+7, 16+koff..}
    const _Float16* pa0 = hin + (long)(b0 + lmod) * HIDDEN + lhalf * 8;
    const _Float16* pa1 = pa0 + (long)16 * HIDDEN;
    // B fragments (W_hh row j, 32x16 f16): lane holds column j, K = {koff..koff+7, 16+koff..}
    const long grow = (long)(j0 + lmod) * HIDDEN + lhalf * 8;
    const _Float16* pbi = whh + grow;
    const _Float16* pbf = whh + grow + (long)2048 * HIDDEN;
    const _Float16* pbg = whh + grow + (long)4096 * HIDDEN;
    const _Float16* pbo = whh + grow + (long)6144 * HIDDEN;

    v8f aI0 = {}, aI1 = {}, aF0 = {}, aF1 = {};
    v8f aG0 = {}, aG1 = {}, aO0 = {}, aO1 = {};

#pragma unroll 2
    for (int k0 = 0; k0 < HIDDEN; k0 += 32) {
        union { v16h v; v8h h[2]; } A0, A1, Bi, Bf, Bg, Bo;
        A0.h[0] = *(const v8h*)(pa0 + k0);
        A0.h[1] = *(const v8h*)(pa0 + k0 + 16);
        A1.h[0] = *(const v8h*)(pa1 + k0);
        A1.h[1] = *(const v8h*)(pa1 + k0 + 16);
        Bi.h[0] = *(const v8h*)(pbi + k0);  Bi.h[1] = *(const v8h*)(pbi + k0 + 16);
        Bf.h[0] = *(const v8h*)(pbf + k0);  Bf.h[1] = *(const v8h*)(pbf + k0 + 16);
        Bg.h[0] = *(const v8h*)(pbg + k0);  Bg.h[1] = *(const v8h*)(pbg + k0 + 16);
        Bo.h[0] = *(const v8h*)(pbo + k0);  Bo.h[1] = *(const v8h*)(pbo + k0 + 16);

        aI0 = __builtin_amdgcn_wmma_f32_16x16x32_f16(false, A0.v, false, Bi.v, (short)0, aI0, false, false);
        aI1 = __builtin_amdgcn_wmma_f32_16x16x32_f16(false, A1.v, false, Bi.v, (short)0, aI1, false, false);
        aF0 = __builtin_amdgcn_wmma_f32_16x16x32_f16(false, A0.v, false, Bf.v, (short)0, aF0, false, false);
        aF1 = __builtin_amdgcn_wmma_f32_16x16x32_f16(false, A1.v, false, Bf.v, (short)0, aF1, false, false);
        aG0 = __builtin_amdgcn_wmma_f32_16x16x32_f16(false, A0.v, false, Bg.v, (short)0, aG0, false, false);
        aG1 = __builtin_amdgcn_wmma_f32_16x16x32_f16(false, A1.v, false, Bg.v, (short)0, aG1, false, false);
        aO0 = __builtin_amdgcn_wmma_f32_16x16x32_f16(false, A0.v, false, Bo.v, (short)0, aO0, false, false);
        aO1 = __builtin_amdgcn_wmma_f32_16x16x32_f16(false, A1.v, false, Bo.v, (short)0, aO1, false, false);
    }

    // Epilogue: D element r of lane = (M = r + 8*lhalf, N = lmod)
    const int j = j0 + lmod;
#pragma unroll
    for (int r = 0; r < 8; ++r) {
        {   // M-tile 0: batch rows b0..b0+15
            int b = b0 + lhalf * 8 + r;
            long gi = (long)b * G4 + j;
            float iv = aI0[r] + xproj[gi];
            float fv = aF0[r] + xproj[gi + 2048];
            float gv = aG0[r] + xproj[gi + 4096];
            float ov = aO0[r] + xproj[gi + 6144];
            long hj = (long)b * HIDDEN + j;
            float si = 1.f / (1.f + __expf(-iv));
            float sf = 1.f / (1.f + __expf(-fv));
            float so = 1.f / (1.f + __expf(-ov));
            float cn = sf * cstate[hj] + si * tanhf(gv);
            cstate[hj] = cn;
            hout[hj] = (_Float16)(so * tanhf(cn));
        }
        {   // M-tile 1: batch rows b0+16..b0+31
            int b = b0 + 16 + lhalf * 8 + r;
            long gi = (long)b * G4 + j;
            float iv = aI1[r] + xproj[gi];
            float fv = aF1[r] + xproj[gi + 2048];
            float gv = aG1[r] + xproj[gi + 4096];
            float ov = aO1[r] + xproj[gi + 6144];
            long hj = (long)b * HIDDEN + j;
            float si = 1.f / (1.f + __expf(-iv));
            float sf = 1.f / (1.f + __expf(-fv));
            float so = 1.f / (1.f + __expf(-ov));
            float cn = sf * cstate[hj] + si * tanhf(gv);
            cstate[hj] = cn;
            hout[hj] = (_Float16)(so * tanhf(cn));
        }
    }
}

// ---------------------------------------------------------------------------
// FC head: out[b, t, o] = h[b,:] . W_fc[o,:] + b_fc[o]
// One wave per (b,o); lane-strided coalesced dot + wave32 shuffle reduction.
// ---------------------------------------------------------------------------
__global__ void fc_kernel(const _Float16* __restrict__ h,
                          const _Float16* __restrict__ wfc,
                          const float*    __restrict__ bfc,
                          float*          __restrict__ out, int t) {
    int wv   = (blockIdx.x * blockDim.x + threadIdx.x) >> 5;
    int lane = threadIdx.x & 31;
    if (wv >= BATCH * OUTD) return;
    int b = wv / OUTD, o = wv % OUTD;
    const _Float16* hr = h   + (long)b * HIDDEN;
    const _Float16* wr = wfc + (long)o * HIDDEN;
    float s = 0.f;
#pragma unroll 4
    for (int k = lane; k < HIDDEN; k += 32)
        s += (float)hr[k] * (float)wr[k];
#pragma unroll
    for (int off = 16; off; off >>= 1) s += __shfl_down(s, off, 32);
    if (lane == 0) out[(long)b * (TFRAME * OUTD) + (long)t * OUTD + o] = s + bfc[o];
}

// ---------------------------------------------------------------------------
// Host launcher
// ---------------------------------------------------------------------------
extern "C" void kernel_launch(void* const* d_in, const int* in_sizes, int n_in,
                              void* d_out, int out_size, void* d_ws, size_t ws_size,
                              hipStream_t stream) {
    const float*     inputs = (const float*)d_in[0];
    const long long* labels = (const long long*)d_in[1];
    const float*     W_ih   = (const float*)d_in[2];
    const float*     W_hh   = (const float*)d_in[3];
    const float*     b_ih   = (const float*)d_in[4];
    const float*     b_hh   = (const float*)d_in[5];
    const float*     W_fc   = (const float*)d_in[6];
    const float*     b_fc   = (const float*)d_in[7];
    float* out = (float*)d_out;

    char* ws = (char*)d_ws;
    size_t off = 0;
    auto alloc = [&](size_t bytes) -> void* {
        void* p = ws + off;
        off += (bytes + 255) & ~(size_t)255;
        return p;
    };
    _Float16* whh16 = (_Float16*)alloc((size_t)G4 * HIDDEN * sizeof(_Float16));   // 32 MB
    _Float16* wfc16 = (_Float16*)alloc((size_t)OUTD * HIDDEN * sizeof(_Float16)); // 264 KB
    float*    xproj = (float*)   alloc((size_t)BATCH * G4 * sizeof(float));       // 8 MB
    _Float16* hbuf0 = (_Float16*)alloc((size_t)BATCH * HIDDEN * sizeof(_Float16));// 1 MB
    _Float16* hbuf1 = (_Float16*)alloc((size_t)BATCH * HIDDEN * sizeof(_Float16));// 1 MB
    float*    cbuf  = (float*)   alloc((size_t)BATCH * HIDDEN * sizeof(float));   // 2 MB

    // One-time precompute (per call; deterministic)
    cvt_f32_f16<<<4096, 256, 0, stream>>>(W_hh, whh16, G4 * HIDDEN);
    cvt_f32_f16<<<132, 256, 0, stream>>>(W_fc, wfc16, OUTD * HIDDEN);
    xproj_kernel<<<(BATCH * G4 + 255) / 256, 256, 0, stream>>>(inputs, labels, W_ih, b_ih, b_hh, xproj);
    init_kernel<<<(BATCH * HIDDEN + 255) / 256, 256, 0, stream>>>(inputs, hbuf0, cbuf, out);

    // 120 sequential LSTM steps (h ping-pong), each followed by the FC head
    _Float16* hin = hbuf0;
    _Float16* hout = hbuf1;
    const int fc_blocks = (BATCH * OUTD * 32 + 255) / 256; // 16896 waves
    for (int s = 0; s < STEPS; ++s) {
        lstm_step<<<256, 128, 0, stream>>>(hin, whh16, xproj, cbuf, hout);
        fc_kernel<<<fc_blocks, 256, 0, stream>>>(hout, wfc16, b_fc, out, s + 1);
        _Float16* tmp = hin; hin = hout; hout = tmp;
    }
}
